// FastMLAttention_4423816314980
// MI455X (gfx1250) — matrix operand order repaired and checked by reading, not compiled, not run on hardware
//
#include <hip/hip_runtime.h>
#include <stdint.h>

// Problem constants (match reference)
#define BB 32
#define LL 512
#define HH 512
#define CC 1000

typedef __attribute__((ext_vector_type(16))) __bf16        v16bf;
typedef __attribute__((ext_vector_type(8)))  float         v8f;
typedef __attribute__((ext_vector_type(4)))  unsigned int  v4u;
typedef __attribute__((ext_vector_type(4)))  float         v4f;

union FragAB { v16bf v; v4u q[2]; unsigned short u[16]; };
union FragC  { v8f   v; float f[8]; };

// fp32 -> bf16, round-to-nearest-even
static __device__ __forceinline__ unsigned short f2bf(float f) {
    unsigned int x = __float_as_uint(f);
    x += 0x7FFFu + ((x >> 16) & 1u);
    return (unsigned short)(x >> 16);
}

// ---------------------------------------------------------------------------
// Prep: inputs f32 [B,L,H]  ->  K16 bf16 [B,L,H]  and  VT16 bf16 [B,H,L]
// (LDS-tiled 32x32 transpose, +1 pad to dodge bank conflicts)
// ---------------------------------------------------------------------------
__global__ __launch_bounds__(256)
void prep_bf16_kernel(const float* __restrict__ inputs,
                      unsigned short* __restrict__ K16,
                      unsigned short* __restrict__ VT16)
{
    __shared__ unsigned short tile[32][33];
    const int b  = blockIdx.z;
    const int l0 = blockIdx.y * 32;
    const int h0 = blockIdx.x * 32;
    const int tc = threadIdx.x & 31;
    const int tr = threadIdx.x >> 5;            // 0..7

    const float* src = inputs + ((size_t)b * LL + l0) * HH + h0;
    #pragma unroll
    for (int i = 0; i < 4; ++i) {
        int r = tr + i * 8;
        unsigned short v = f2bf(src[(size_t)r * HH + tc]);
        K16[((size_t)b * LL + l0 + r) * HH + h0 + tc] = v;
        tile[r][tc] = v;
    }
    __syncthreads();
    #pragma unroll
    for (int i = 0; i < 4; ++i) {
        int r = tr + i * 8;
        VT16[((size_t)b * HH + h0 + r) * LL + l0 + tc] = tile[tc][r];
    }
}

// ---------------------------------------------------------------------------
// Main attention kernel. USE_WS=true: bf16 operands from workspace (b128
// fragment loads, no conversions in the GEMM loops). USE_WS=false: fallback
// converting f32 on the fly.
// ---------------------------------------------------------------------------
template <bool USE_WS>
__global__ __launch_bounds__(256)
void attn_wmma_kernel(const float*   __restrict__ inputs,   // [B,L,H] f32
                      const uint8_t* __restrict__ masks,    // [B,L] bool
                      const int*     __restrict__ cand,     // [B,C] i32
                      const float*   __restrict__ attn_emb, // [V,H] f32
                      float*         __restrict__ out,      // [B,C,H] f32
                      const unsigned short* __restrict__ K16,   // [B,L,H] bf16
                      const unsigned short* __restrict__ VT16)  // [B,H,L] bf16
{
    extern __shared__ unsigned char smem[];
    // LDS layout (96 KB): Q bf16[32][512] (32 KB, dead after GEMM1),
    //                     S f32 [32][512] (64 KB), P bf16 overlays Q.
    unsigned short* Q = (unsigned short*)smem;
    float*          S = (float*)(smem + 32 * HH * 2);
    unsigned short* P = (unsigned short*)smem;

    const int cb   = blockIdx.x;
    const int b    = blockIdx.y;
    const int tid  = threadIdx.x;
    const int wave = tid >> 5;
    const int lane = tid & 31;
    const int hl   = lane >> 4;        // lane half
    const int ln   = lane & 15;

    const float* inB = inputs + (size_t)b * LL * HH;

    // ---------------- Phase 0: gather Q (fp32 -> bf16 into LDS) -------------
    for (int idx = tid; idx < 32 * (HH / 4); idx += 256) {
        int r  = idx >> 7;                    // HH/4 == 128
        int h4 = (idx & 127) * 4;
        int cidx = cb * 32 + r;
        int vid  = (cidx < CC) ? cand[(size_t)b * CC + cidx] : 0;
        v4f f = *(const v4f*)(attn_emb + (size_t)vid * HH + h4);
        unsigned short* dst = Q + r * HH + h4;
        dst[0] = f2bf(f.x); dst[1] = f2bf(f.y);
        dst[2] = f2bf(f.z); dst[3] = f2bf(f.w);
    }
    __syncthreads();

    // ---------------- Phase 1: S = Q * K^T  (m=c, n=l, k=h) -----------------
    for (int Nj = wave; Nj < 32; Nj += 8) {
        const int l0 = Nj * 16;
        FragC acc0 = {}, acc1 = {};
        for (int kk = 0; kk < HH; kk += 32) {
            FragAB fb;
            if constexpr (USE_WS) {
                const unsigned short* kb =
                    K16 + ((size_t)b * LL + l0 + ln) * HH + kk + hl * 16;
                fb.q[0] = *(const v4u*)(kb);
                fb.q[1] = *(const v4u*)(kb + 8);
            } else {
                const float* src = inB + (size_t)(l0 + ln) * HH + kk + hl * 16;
                #pragma unroll
                for (int e = 0; e < 16; ++e) fb.u[e] = f2bf(src[e]);
            }
            const int khalf = hl * 16;
            const unsigned short* qa = Q + (size_t)ln        * HH + kk + khalf;
            const unsigned short* qb = Q + (size_t)(16 + ln) * HH + kk + khalf;
            FragAB fa0, fa1;
            fa0.q[0] = *(const v4u*)(qa);
            fa0.q[1] = *(const v4u*)(qa + 8);
            fa1.q[0] = *(const v4u*)(qb);
            fa1.q[1] = *(const v4u*)(qb + 8);

            acc0.v = __builtin_amdgcn_wmma_f32_16x16x32_bf16(
                false, fa0.v, false, fb.v, (short)0, acc0.v, false, false);
            acc1.v = __builtin_amdgcn_wmma_f32_16x16x32_bf16(
                false, fa1.v, false, fb.v, (short)0, acc1.v, false, false);
        }
        #pragma unroll
        for (int r = 0; r < 8; ++r) {
            int m = hl * 8 + r;
            S[(size_t)m        * LL + l0 + ln] = acc0.f[r];
            S[(size_t)(16 + m) * LL + l0 + ln] = acc1.f[r];
        }
    }
    __syncthreads();

    // ---------------- Phase 2: masked softmax over L, P = bf16 --------------
    {
        const int row = tid >> 3;
        const int sub = tid & 7;
        const uint8_t* mrow = masks + (size_t)b * LL;
        float* srow = S + (size_t)row * LL;

        float mx = -__builtin_inff();
        for (int l = sub; l < LL; l += 8) {
            float s = mrow[l] ? srow[l] : -__builtin_inff();
            srow[l] = s;
            mx = fmaxf(mx, s);
        }
        mx = fmaxf(mx, __shfl_xor(mx, 1, 32));
        mx = fmaxf(mx, __shfl_xor(mx, 2, 32));
        mx = fmaxf(mx, __shfl_xor(mx, 4, 32));

        float sum = 0.0f;
        for (int l = sub; l < LL; l += 8) {
            float e = __expf(srow[l] - mx);
            srow[l] = e;
            sum += e;
        }
        sum += __shfl_xor(sum, 1, 32);
        sum += __shfl_xor(sum, 2, 32);
        sum += __shfl_xor(sum, 4, 32);
        const float inv = 1.0f / sum;

        for (int l = sub; l < LL; l += 8)
            P[(size_t)row * LL + l] = f2bf(srow[l] * inv);
    }
    __syncthreads();

    // ---------------- Phase 3: O = P * V  (m=c, n=h, k=l) -------------------
    for (int Hj = wave; Hj < 32; Hj += 8) {
        const int h0 = Hj * 16;
        FragC acc0 = {}, acc1 = {};
        for (int kk = 0; kk < LL; kk += 32) {
            FragAB fb;
            if constexpr (USE_WS) {
                const unsigned short* vb =
                    VT16 + ((size_t)b * HH + h0 + ln) * LL + kk + hl * 16;
                __builtin_prefetch(vb + 32, 0, 0);   // next k-tile
                fb.q[0] = *(const v4u*)(vb);
                fb.q[1] = *(const v4u*)(vb + 8);
            } else {
                const int kbase = kk + hl * 16;
                const float* src = inB + (size_t)kbase * HH + h0 + ln;
                __builtin_prefetch(src + (size_t)32 * HH, 0, 0);
                #pragma unroll
                for (int e = 0; e < 16; ++e)
                    fb.u[e] = f2bf(src[(size_t)e * HH]);
            }
            const int khalf = hl * 16;
            const unsigned short* pa = P + (size_t)ln        * LL + kk + khalf;
            const unsigned short* pb = P + (size_t)(16 + ln) * LL + kk + khalf;
            FragAB fa0, fa1;
            fa0.q[0] = *(const v4u*)(pa);
            fa0.q[1] = *(const v4u*)(pa + 8);
            fa1.q[0] = *(const v4u*)(pb);
            fa1.q[1] = *(const v4u*)(pb + 8);

            acc0.v = __builtin_amdgcn_wmma_f32_16x16x32_bf16(
                false, fa0.v, false, fb.v, (short)0, acc0.v, false, false);
            acc1.v = __builtin_amdgcn_wmma_f32_16x16x32_bf16(
                false, fa1.v, false, fb.v, (short)0, acc1.v, false, false);
        }
        #pragma unroll
        for (int r = 0; r < 8; ++r) {
            int m  = hl * 8 + r;
            int c0 = cb * 32 + m;
            int c1 = cb * 32 + 16 + m;
            if (c0 < CC) out[((size_t)b * CC + c0) * HH + h0 + ln] = acc0.f[r];
            if (c1 < CC) out[((size_t)b * CC + c1) * HH + h0 + ln] = acc1.f[r];
        }
    }
}

extern "C" void kernel_launch(void* const* d_in, const int* in_sizes, int n_in,
                              void* d_out, int out_size, void* d_ws, size_t ws_size,
                              hipStream_t stream) {
    (void)in_sizes; (void)n_in; (void)out_size;
    const float*   inputs   = (const float*)  d_in[0];
    const uint8_t* masks    = (const uint8_t*)d_in[1];
    const int*     cand     = (const int*)    d_in[2];
    const float*   attn_emb = (const float*)  d_in[3];
    float*         out      = (float*)        d_out;

    dim3 grid((CC + 31) / 32, BB);           // 32 x 32 = 1024 workgroups
    dim3 block(256);                          // 8 wave32 waves
    size_t shmem = (size_t)32 * HH * 2        // Q bf16 / P overlay
                 + (size_t)32 * LL * 4;       // S f32   -> 96 KB

    const size_t need = (size_t)2 * BB * LL * HH * 2;   // K16 + VT16 = 32 MB
    if (d_ws != nullptr && ws_size >= need) {
        unsigned short* K16  = (unsigned short*)d_ws;
        unsigned short* VT16 = K16 + (size_t)BB * LL * HH;
        dim3 pgrid(HH / 32, LL / 32, BB);
        hipLaunchKernelGGL(prep_bf16_kernel, pgrid, dim3(256), 0, stream,
                           inputs, K16, VT16);
        hipLaunchKernelGGL((attn_wmma_kernel<true>), grid, block, shmem, stream,
                           inputs, masks, cand, attn_emb, out, K16, VT16);
    } else {
        hipLaunchKernelGGL((attn_wmma_kernel<false>), grid, block, shmem, stream,
                           inputs, masks, cand, attn_emb, out, nullptr, nullptr);
    }
}
